// MambaHAR_88854283419802
// MI455X (gfx1250) — compile-verified, matrix operand order checked by reading, not run on hardware
//
#include <hip/hip_runtime.h>
#include <hip/hip_bf16.h>

typedef __attribute__((ext_vector_type(16))) _Float16 v16h;
typedef __attribute__((ext_vector_type(8)))  _Float16 v8h;
typedef __attribute__((ext_vector_type(8)))  float    v8f;

constexpr int B_  = 32, L_ = 1024, DIN_ = 9, DM_ = 512, NL_ = 4, DI_ = 1024;
constexpr int DS_ = 16, DC_ = 4, DTR_ = 32, NC_ = 12;
constexpr int BL_ = B_ * L_;          // 32768 rows
constexpr int E2_ = 2 * DI_;          // 2048
constexpr int XD_ = DTR_ + 2 * DS_;   // 64

// ---------------------------------------------------------------------------
// f16 WMMA GEMM:  C[M,N] = A[M,K] * W[N,K]^T   (wave tile 32x64: 2x4 WMMAs)
// epilogue: 0 = f32 store, 1 = f16 store, 2 = softplus(acc+aux[col]) f32,
//           3 = acc + res[idx] -> f32
// ---------------------------------------------------------------------------
__global__ void __launch_bounds__(128)
wmma_gemm_kernel(const _Float16* __restrict__ A, const _Float16* __restrict__ W,
                 float* Cf, _Float16* Ch,
                 int M, int N, int K, int epilogue,
                 const float* __restrict__ aux, const float* res)
{
    const int lane = threadIdx.x & 31;
    const int wid  = threadIdx.x >> 5;
    const int nTilesN = N >> 6;
    const long gw = (long)blockIdx.x * 4 + wid;
    const long totalTiles = (long)(M >> 5) * nTilesN;
    if (gw >= totalTiles) return;
    const int m0 = ((int)(gw / nTilesN)) << 5;
    const int n0 = ((int)(gw % nTilesN)) << 6;
    const int r    = lane & 15;
    const int half = lane >> 4;

    // Per documented 16-bit A layout: lanes 0-15 hold K 0-7/16-23, lanes 16-31
    // hold K 8-15/24-31; B fragments are weight rows (B = W^T columns).
    const _Float16* A0 = A + (size_t)(m0 +  0 + r) * K + half * 8;
    const _Float16* A1 = A + (size_t)(m0 + 16 + r) * K + half * 8;
    const _Float16* W0 = W + (size_t)(n0 +  0 + r) * K + half * 8;
    const _Float16* W1 = W + (size_t)(n0 + 16 + r) * K + half * 8;
    const _Float16* W2 = W + (size_t)(n0 + 32 + r) * K + half * 8;
    const _Float16* W3 = W + (size_t)(n0 + 48 + r) * K + half * 8;

    union V16 { v16h v; v8h h[2]; };
    v8f c00 = {}, c01 = {}, c02 = {}, c03 = {};
    v8f c10 = {}, c11 = {}, c12 = {}, c13 = {};

    for (int kb = 0; kb < K; kb += 32) {
        V16 a0, a1, b0, b1, b2, b3;
        a0.h[0] = *(const v8h*)(A0 + kb);  a0.h[1] = *(const v8h*)(A0 + kb + 16);
        a1.h[0] = *(const v8h*)(A1 + kb);  a1.h[1] = *(const v8h*)(A1 + kb + 16);
        b0.h[0] = *(const v8h*)(W0 + kb);  b0.h[1] = *(const v8h*)(W0 + kb + 16);
        b1.h[0] = *(const v8h*)(W1 + kb);  b1.h[1] = *(const v8h*)(W1 + kb + 16);
        b2.h[0] = *(const v8h*)(W2 + kb);  b2.h[1] = *(const v8h*)(W2 + kb + 16);
        b3.h[0] = *(const v8h*)(W3 + kb);  b3.h[1] = *(const v8h*)(W3 + kb + 16);
        if (kb + 32 < K) {                    // emits global_prefetch_b8
            __builtin_prefetch((const void*)(A0 + kb + 32), 0, 1);
            __builtin_prefetch((const void*)(A1 + kb + 32), 0, 1);
            __builtin_prefetch((const void*)(W0 + kb + 32), 0, 1);
            __builtin_prefetch((const void*)(W2 + kb + 32), 0, 1);
        }
        c00 = __builtin_amdgcn_wmma_f32_16x16x32_f16(false, a0.v, false, b0.v, (short)0, c00, false, false);
        c01 = __builtin_amdgcn_wmma_f32_16x16x32_f16(false, a0.v, false, b1.v, (short)0, c01, false, false);
        c02 = __builtin_amdgcn_wmma_f32_16x16x32_f16(false, a0.v, false, b2.v, (short)0, c02, false, false);
        c03 = __builtin_amdgcn_wmma_f32_16x16x32_f16(false, a0.v, false, b3.v, (short)0, c03, false, false);
        c10 = __builtin_amdgcn_wmma_f32_16x16x32_f16(false, a1.v, false, b0.v, (short)0, c10, false, false);
        c11 = __builtin_amdgcn_wmma_f32_16x16x32_f16(false, a1.v, false, b1.v, (short)0, c11, false, false);
        c12 = __builtin_amdgcn_wmma_f32_16x16x32_f16(false, a1.v, false, b2.v, (short)0, c12, false, false);
        c13 = __builtin_amdgcn_wmma_f32_16x16x32_f16(false, a1.v, false, b3.v, (short)0, c13, false, false);
    }

    v8f acc[2][4] = {{c00, c01, c02, c03}, {c10, c11, c12, c13}};
    const int colb = lane & 15;
#pragma unroll
    for (int mi = 0; mi < 2; ++mi) {
        const int rowb = m0 + mi * 16 + half * 8;
#pragma unroll
        for (int ni = 0; ni < 4; ++ni) {
            const int col = n0 + ni * 16 + colb;
#pragma unroll
            for (int rr = 0; rr < 8; ++rr) {
                size_t idx = (size_t)(rowb + rr) * N + col;
                float v = acc[mi][ni][rr];
                if (epilogue == 2) { v += aux[col]; v = (v > 20.f) ? v : log1pf(__expf(v)); }
                else if (epilogue == 3) { v += res[idx]; }
                if (epilogue == 1) Ch[idx] = (_Float16)v;
                else               Cf[idx] = v;
            }
        }
    }
}

// ---------------------------------------------------------------------------
__global__ void cvt_f16_kernel(const float* __restrict__ in, _Float16* __restrict__ out, long n)
{
    long i = (long)blockIdx.x * blockDim.x + threadIdx.x;
    if (i < n) out[i] = (_Float16)in[i];
}

__global__ void negexp_kernel(const float* __restrict__ in, float* __restrict__ out, long n)
{
    long i = (long)blockIdx.x * blockDim.x + threadIdx.x;
    if (i < n) out[i] = -__expf(in[i]);
}

__global__ void input_proj_kernel(const float* __restrict__ x, const float* __restrict__ w,
                                  const float* __restrict__ bias, float* __restrict__ h)
{
    long i = (long)blockIdx.x * blockDim.x + threadIdx.x;
    if (i >= (long)BL_ * DM_) return;
    const long rowl = i / DM_;
    const int  d    = (int)(i % DM_);
    const float* xr = x + rowl * DIN_;
    const float* wr = w + (size_t)d * DIN_;
    float s = bias[d];
#pragma unroll
    for (int k = 0; k < DIN_; ++k) s += xr[k] * wr[k];
    h[i] = s;
}

__global__ void __launch_bounds__(256)
ln_kernel(const float* __restrict__ h, const float* __restrict__ g,
          const float* __restrict__ bb, _Float16* __restrict__ out)
{
    const int row = blockIdx.x;
    const int tid = threadIdx.x;
    const float* x = h + (size_t)row * DM_;
    float v0 = x[tid], v1 = x[tid + 256];
    __shared__ float s1[256], s2[256];
    s1[tid] = v0 + v1;
    s2[tid] = v0 * v0 + v1 * v1;
    __syncthreads();
    for (int off = 128; off > 0; off >>= 1) {
        if (tid < off) { s1[tid] += s1[tid + off]; s2[tid] += s2[tid + off]; }
        __syncthreads();
    }
    const float mean = s1[0] * (1.f / DM_);
    const float var  = s2[0] * (1.f / DM_) - mean * mean;
    const float inv  = rsqrtf(var + 1e-5f);
    out[(size_t)row * DM_ + tid]       = (_Float16)((v0 - mean) * inv * g[tid] + bb[tid]);
    out[(size_t)row * DM_ + tid + 256] = (_Float16)((v1 - mean) * inv * g[tid + 256] + bb[tid + 256]);
}

// causal depthwise conv (DC=4) + SiLU ; reads u-half of xz16, writes u16
__global__ void conv_silu_kernel(const _Float16* __restrict__ xz16,
                                 const float* __restrict__ cw, const float* __restrict__ cb,
                                 _Float16* __restrict__ u16)
{
    long i = (long)blockIdx.x * blockDim.x + threadIdx.x;
    if (i >= (long)BL_ * DI_) return;
    const int  e    = (int)(i % DI_);
    const long rowl = i / DI_;          // b*L + l
    const int  l    = (int)(rowl % L_);
    float acc = cb[e];
    const float* w = cw + (size_t)e * DC_;
#pragma unroll
    for (int j = 0; j < DC_; ++j) {
        const int l2 = l + j - (DC_ - 1);
        if (l2 >= 0)
            acc += w[j] * (float)xz16[(rowl + (long)(l2 - l)) * E2_ + e];
    }
    const float s = acc / (1.f + __expf(-acc));   // silu
    u16[i] = (_Float16)s;
}

__global__ void dtcvt_kernel(const float* __restrict__ xdbl, _Float16* __restrict__ dt16)
{
    long i = (long)blockIdx.x * blockDim.x + threadIdx.x;
    if (i >= (long)BL_ * DTR_) return;
    const long rowl = i / DTR_;
    const int  c    = (int)(i % DTR_);
    dt16[i] = (_Float16)xdbl[rowl * XD_ + c];
}

// selective scan: one thread per (b, e); B/C staged per-timestep via
// double-buffered LDS (single barrier per step).
__global__ void __launch_bounds__(256)
scan_kernel(const float* __restrict__ delta, const _Float16* __restrict__ u16,
            const _Float16* __restrict__ xz16, const float* __restrict__ xdbl,
            const float* __restrict__ negA, const float* __restrict__ Dp,
            _Float16* __restrict__ y16)
{
    const int b = blockIdx.y;
    const int e = blockIdx.x * 256 + threadIdx.x;
    __shared__ float sBC[2][32];
    float hs[DS_], An[DS_];
#pragma unroll
    for (int s = 0; s < DS_; ++s) { hs[s] = 0.f; An[s] = negA[(size_t)e * DS_ + s]; }
    const float dcoef = Dp[e];
    for (int l = 0; l < L_; ++l) {
        const size_t row = (size_t)b * L_ + l;
        const int buf = l & 1;
        if (threadIdx.x < 32) sBC[buf][threadIdx.x] = xdbl[row * XD_ + DTR_ + threadIdx.x];
        __syncthreads();
        const float d  = delta[row * DI_ + e];
        const float uv = (float)u16[row * DI_ + e];
        const float zv = (float)xz16[row * E2_ + DI_ + e];
        const float du = d * uv;
        float y = 0.f;
#pragma unroll
        for (int s = 0; s < DS_; ++s) {
            const float dA = __expf(d * An[s]);
            hs[s] = dA * hs[s] + du * sBC[buf][s];
            y += hs[s] * sBC[buf][16 + s];
        }
        y += uv * dcoef;
        y *= zv / (1.f + __expf(-zv));            // * silu(z)
        y16[row * DI_ + e] = (_Float16)y;
    }
}

__global__ void __launch_bounds__(256)
pool_fc_kernel(const float* __restrict__ h, const float* __restrict__ fcw,
               const float* __restrict__ fcb, float* __restrict__ out)
{
    const int b = blockIdx.x;
    const int tid = threadIdx.x;
    __shared__ float pooled[DM_];
    for (int d = tid; d < DM_; d += 256) {
        float s = 0.f;
        const float* p = h + (size_t)b * L_ * DM_ + d;
        for (int l = 0; l < L_; ++l) s += p[(size_t)l * DM_];
        pooled[d] = s * (1.f / L_);
    }
    __syncthreads();
    if (tid < NC_) {
        float s = fcb[tid];
        const float* w = fcw + (size_t)tid * DM_;
        for (int d = 0; d < DM_; ++d) s += pooled[d] * w[d];
        out[b * NC_ + tid] = s;
    }
}

// ---------------------------------------------------------------------------
extern "C" void kernel_launch(void* const* d_in, const int* in_sizes, int n_in,
                              void* d_out, int out_size, void* d_ws, size_t ws_size,
                              hipStream_t stream)
{
    const float* x      = (const float*)d_in[0];
    const float* ip_w   = (const float*)d_in[1];
    const float* ip_b   = (const float*)d_in[2];
    const float* ln_g   = (const float*)d_in[3];
    const float* ln_b   = (const float*)d_in[4];
    const float* in_w   = (const float*)d_in[5];
    const float* conv_w = (const float*)d_in[6];
    const float* conv_b = (const float*)d_in[7];
    const float* xp_w   = (const float*)d_in[8];
    const float* dt_w   = (const float*)d_in[9];
    const float* dt_b   = (const float*)d_in[10];
    const float* A_log  = (const float*)d_in[11];
    const float* Dvec   = (const float*)d_in[12];
    const float* out_w  = (const float*)d_in[13];
    const float* fc_w   = (const float*)d_in[14];
    const float* fc_b   = (const float*)d_in[15];
    float* out = (float*)d_out;
    (void)in_sizes; (void)n_in; (void)out_size; (void)ws_size;

    char* wsb = (char*)d_ws;
    size_t off = 0;
    auto alloc = [&](size_t bytes) -> char* {
        off = (off + 255) & ~(size_t)255;
        char* p = wsb + off; off += bytes; return p;
    };
    _Float16* w16_in  = (_Float16*)alloc((size_t)NL_ * E2_ * DM_ * 2);
    _Float16* w16_xp  = (_Float16*)alloc((size_t)NL_ * XD_ * DI_ * 2);
    _Float16* w16_dt  = (_Float16*)alloc((size_t)NL_ * DI_ * DTR_ * 2);
    _Float16* w16_out = (_Float16*)alloc((size_t)NL_ * DM_ * DI_ * 2);
    float*    negA    = (float*)   alloc((size_t)NL_ * DI_ * DS_ * 4);
    float*    hbuf    = (float*)   alloc((size_t)BL_ * DM_ * 4);
    _Float16* hn16    = (_Float16*)alloc((size_t)BL_ * DM_ * 2);
    _Float16* xz16    = (_Float16*)alloc((size_t)BL_ * E2_ * 2);
    _Float16* u16     = (_Float16*)alloc((size_t)BL_ * DI_ * 2);
    float*    xdbl    = (float*)   alloc((size_t)BL_ * XD_ * 4);
    _Float16* dt16    = (_Float16*)alloc((size_t)BL_ * DTR_ * 2);
    float*    delta   = (float*)   alloc((size_t)BL_ * DI_ * 4);
    _Float16* y16     = (_Float16*)alloc((size_t)BL_ * DI_ * 2);

    auto cvt = [&](const float* src, _Float16* dst, long n) {
        cvt_f16_kernel<<<(n + 255) / 256, 256, 0, stream>>>(src, dst, n);
    };
    cvt(in_w,  w16_in,  (long)NL_ * E2_ * DM_);
    cvt(xp_w,  w16_xp,  (long)NL_ * XD_ * DI_);
    cvt(dt_w,  w16_dt,  (long)NL_ * DI_ * DTR_);
    cvt(out_w, w16_out, (long)NL_ * DM_ * DI_);
    { long n = (long)NL_ * DI_ * DS_;
      negexp_kernel<<<(n + 255) / 256, 256, 0, stream>>>(A_log, negA, n); }

    { long n = (long)BL_ * DM_;
      input_proj_kernel<<<(n + 255) / 256, 256, 0, stream>>>(x, ip_w, ip_b, hbuf); }

    auto gemm = [&](const _Float16* A, const _Float16* W, float* Cf, _Float16* Ch,
                    int M, int N, int K, int ep, const float* aux, const float* res) {
        long tiles  = (long)(M / 32) * (N / 64);
        long blocks = (tiles + 3) / 4;
        wmma_gemm_kernel<<<blocks, 128, 0, stream>>>(A, W, Cf, Ch, M, N, K, ep, aux, res);
    };

    for (int i = 0; i < NL_; ++i) {
        ln_kernel<<<BL_, 256, 0, stream>>>(hbuf, ln_g + (size_t)i * DM_, ln_b + (size_t)i * DM_, hn16);
        gemm(hn16, w16_in + (size_t)i * E2_ * DM_, nullptr, xz16, BL_, E2_, DM_, 1, nullptr, nullptr);
        { long n = (long)BL_ * DI_;
          conv_silu_kernel<<<(n + 255) / 256, 256, 0, stream>>>(
              xz16, conv_w + (size_t)i * DI_ * DC_, conv_b + (size_t)i * DI_, u16); }
        gemm(u16, w16_xp + (size_t)i * XD_ * DI_, xdbl, nullptr, BL_, XD_, DI_, 0, nullptr, nullptr);
        { long n = (long)BL_ * DTR_;
          dtcvt_kernel<<<(n + 255) / 256, 256, 0, stream>>>(xdbl, dt16); }
        gemm(dt16, w16_dt + (size_t)i * DI_ * DTR_, delta, nullptr, BL_, DI_, DTR_, 2,
             dt_b + (size_t)i * DI_, nullptr);
        scan_kernel<<<dim3(DI_ / 256, B_), 256, 0, stream>>>(
            delta, u16, xz16, xdbl, negA + (size_t)i * DI_ * DS_, Dvec + (size_t)i * DI_, y16);
        gemm(y16, w16_out + (size_t)i * DM_ * DI_, hbuf, nullptr, BL_, DM_, DI_, 3, nullptr, hbuf);
    }

    pool_fc_kernel<<<B_, 256, 0, stream>>>(hbuf, fc_w, fc_b, out);
}
